// FastHybridQuantumAttention_38826504356542
// MI455X (gfx1250) — matrix operand order repaired and checked by reading, not compiled, not run on hardware
//
#include <hip/hip_runtime.h>
#include <hip/hip_bf16.h>
#include <math.h>

// ---------------------------------------------------------------------------
// Types for CDNA5 WMMA (wave32): v_wmma_f32_16x16x32_bf16
// ---------------------------------------------------------------------------
typedef __attribute__((ext_vector_type(16))) __bf16 v16bf;
typedef __attribute__((ext_vector_type(8)))  __bf16 bf16x8;
typedef __attribute__((ext_vector_type(8)))  float  v8f;

union BF16x16 { v16bf v; bf16x8 h[2]; };

static __device__ inline v8f wmma_bf16(v16bf a, v16bf b, v8f c) {
    // D = A x B + C, bf16 inputs, f32 accumulate
    return __builtin_amdgcn_wmma_f32_16x16x32_bf16(
        /*neg_a=*/false, a, /*neg_b=*/false, b,
        /*c_mod=*/(short)0, c, /*reuse_a=*/false, /*reuse_b=*/false);
}

// Build A-operand (16x32 bf16, M=lane%16, K window part*16 .. part*16+15)
static __device__ inline v16bf ld_a2(const __bf16* plo, const __bf16* phi) {
    BF16x16 u;
    u.h[0] = *(const bf16x8*)plo;
    u.h[1] = *(const bf16x8*)phi;
    return u.v;
}
static __device__ inline v16bf ld_b16(const __bf16* p) {
    return *(const v16bf*)p;
}

// ---------------------------------------------------------------------------
// Problem constants
// ---------------------------------------------------------------------------
#define S_LEN     1024
#define DIMM      1024
#define HEADD     64
#define NHEADS    16
#define QHEADS    2
#define THEADS    14
#define KSTRIDE   128     // per-head Q/K feature row stride
#define SCALE_ATT 0.125f  // 1/sqrt(64)

// ---------------------------------------------------------------------------
// Kernel: f32 -> bf16 conversion
// ---------------------------------------------------------------------------
__global__ void cvt_f32_bf16(const float* __restrict__ in, __bf16* __restrict__ out, int n) {
    int i = blockIdx.x * blockDim.x + threadIdx.x;
    if (i < n) out[i] = (__bf16)in[i];
}

// ---------------------------------------------------------------------------
// Kernel: quantum head preprocessing.
// qs = tanh(LN(xh @ qsW^T + qsb)); amp=qs[:64], phase=qs[64:]
// Q-feat = [amp*cos(f*ph+s), amp*sin(f*ph+s)] * scale
// K-feat = [amp*cos(f*ph),   amp*sin(f*ph)]
// Vt[h][d][s] = x[s][h*64+d]  (transposed value matrix, bf16)
// grid = (S, QHEADS), block = 128
// ---------------------------------------------------------------------------
__global__ __launch_bounds__(128) void quantum_prep(
    const float* __restrict__ x, const float* __restrict__ qsW,
    const float* __restrict__ qsb, const float* __restrict__ lng,
    const float* __restrict__ lnb, const float* __restrict__ qfreq,
    const float* __restrict__ qshift,
    __bf16* __restrict__ Qb, __bf16* __restrict__ Kb, __bf16* __restrict__ Vt)
{
    const int s = blockIdx.x;
    const int h = blockIdx.y;
    const int t = threadIdx.x;

    __shared__ float xh[64];
    __shared__ float qsv[128];
    __shared__ float red[128];

    if (t < 64) xh[t] = x[(size_t)s * DIMM + h * HEADD + t];
    __syncthreads();

    float acc = qsb[t];
    const float* wrow = qsW + (size_t)t * 64;
    #pragma unroll 8
    for (int k = 0; k < 64; ++k) acc += xh[k] * wrow[k];

    // layernorm over 128 features
    red[t] = acc; __syncthreads();
    for (int st = 64; st > 0; st >>= 1) {
        if (t < st) red[t] += red[t + st];
        __syncthreads();
    }
    float mu = red[0] * (1.0f / 128.0f);
    __syncthreads();
    float d = acc - mu;
    red[t] = d * d; __syncthreads();
    for (int st = 64; st > 0; st >>= 1) {
        if (t < st) red[t] += red[t + st];
        __syncthreads();
    }
    float var = red[0] * (1.0f / 128.0f);
    __syncthreads();

    float y = d * rsqrtf(var + 1e-5f) * lng[t] + lnb[t];
    qsv[t] = tanhf(y);
    __syncthreads();

    if (t < 64) {
        float amp = qsv[t];
        float ph  = qsv[64 + t];
        float f   = qfreq[h];
        float sh  = qshift[h];
        float a1  = f * ph + sh;  // query-side angle
        float a2  = f * ph;       // key-side angle
        __bf16* qrow = Qb + (size_t)h * S_LEN * KSTRIDE + (size_t)s * KSTRIDE;
        __bf16* krow = Kb + (size_t)h * S_LEN * KSTRIDE + (size_t)s * KSTRIDE;
        qrow[t]      = (__bf16)(amp * __cosf(a1) * SCALE_ATT);
        qrow[64 + t] = (__bf16)(amp * __sinf(a1) * SCALE_ATT);
        krow[t]      = (__bf16)(amp * __cosf(a2));
        krow[64 + t] = (__bf16)(amp * __sinf(a2));
        Vt[(size_t)h * HEADD * S_LEN + (size_t)t * S_LEN + s] = (__bf16)xh[t];
    }
}

// ---------------------------------------------------------------------------
// bf16 GEMM: C[M,N] = A[MxK] * Bm^T  (Bm is [N x K] row-major).
// One wave per 32x64 output tile (2 M sub-tiles share each B load).
// grid = (M/32, N/64).
// Epilogue modes:
//   5: packed QKV epilogue; merged B = [Wq;Wk;Wv] (N=2688).
//      th = col/64: th<14 -> Qpack (scaled), th<28 -> Kpack, else Vt pack.
//   4: of[row*ldo + col] = v + bias[col]  (final f32 output)
// ---------------------------------------------------------------------------
__global__ __launch_bounds__(32) void gemm_bf16_nt(
    const __bf16* __restrict__ A, const __bf16* __restrict__ Bm,
    int K, int mode, const float* __restrict__ bias,
    __bf16* __restrict__ pQ, __bf16* __restrict__ pK, __bf16* __restrict__ pV,
    float* __restrict__ of, int ldo)
{
    const int mt   = blockIdx.x;
    const int nt   = blockIdx.y;
    const int lane = threadIdx.x;
    const int col  = lane & 15;
    const int part = lane >> 4;

    const __bf16* arow0 = A + (size_t)(mt * 32 + col) * K;
    const __bf16* arow1 = arow0 + (size_t)16 * K;
    v8f c[2][4] = {};

    for (int kc = 0; kc < K; kc += 32) {
        v16bf a0 = ld_a2(arow0 + kc + part * 16, arow0 + kc + part * 16 + 8);
        v16bf a1 = ld_a2(arow1 + kc + part * 16, arow1 + kc + part * 16 + 8);
        #pragma unroll
        for (int j = 0; j < 4; ++j) {
            const __bf16* brow =
                Bm + (size_t)(nt * 64 + j * 16 + col) * K + kc + part * 16;
            v16bf b = ld_b16(brow);
            c[0][j] = wmma_bf16(a0, b, c[0][j]);
            c[1][j] = wmma_bf16(a1, b, c[1][j]);
        }
    }

    #pragma unroll
    for (int ms = 0; ms < 2; ++ms) {
        #pragma unroll
        for (int j = 0; j < 4; ++j) {
            #pragma unroll
            for (int i = 0; i < 8; ++i) {
                int row  = mt * 32 + ms * 16 + part * 8 + i;
                int ocol = nt * 64 + j * 16 + col;
                float v  = c[ms][j][i];
                if (mode == 5) {
                    int th = ocol >> 6, dd = ocol & 63;
                    if (th < THEADS) {
                        pQ[(size_t)th * S_LEN * KSTRIDE + (size_t)row * KSTRIDE + dd] =
                            (__bf16)(v * SCALE_ATT);
                    } else if (th < 2 * THEADS) {
                        pK[(size_t)(th - THEADS) * S_LEN * KSTRIDE +
                           (size_t)row * KSTRIDE + dd] = (__bf16)v;
                    } else {
                        pV[(size_t)(th - 2 * THEADS) * HEADD * S_LEN +
                           (size_t)dd * S_LEN + row] = (__bf16)v;
                    }
                } else {
                    of[(size_t)row * ldo + ocol] = v + bias[ocol];
                }
            }
        }
    }
}

// ---------------------------------------------------------------------------
// Flash attention. One wave per (head, 32-query tile): 2 query sub-tiles
// share every K-tile and V-tile load. NA = kdim/32 (4 quantum, 2 trad).
// Qb/Kb rows strided KSTRIDE; Vt is [head][64][S] (transposed values).
// Output written as bf16 into concat[s][head*64 + d].
// grid = (S/32, nheads), block = 32
// ---------------------------------------------------------------------------
template <int NA>
__global__ __launch_bounds__(32) void flash_attn(
    const __bf16* __restrict__ Qb, const __bf16* __restrict__ Kb,
    const __bf16* __restrict__ Vt, __bf16* __restrict__ concat, int headOfs)
{
    const int head = blockIdx.y + headOfs;
    const int qt   = blockIdx.x;
    const int lane = threadIdx.x;
    const int col  = lane & 15;
    const int part = lane >> 4;

    __shared__ __align__(32) __bf16 pl[32][32];

    // Preload both Q sub-tiles (A layout); scale folded in.
    v16bf aq[2][NA];
    #pragma unroll
    for (int q = 0; q < 2; ++q) {
        const __bf16* qrow = Qb + (size_t)head * S_LEN * KSTRIDE +
                             (size_t)(qt * 32 + q * 16 + col) * KSTRIDE;
        #pragma unroll
        for (int c = 0; c < NA; ++c) {
            int kc = c * 32;
            aq[q][c] = ld_a2(qrow + kc + part * 16, qrow + kc + part * 16 + 8);
        }
    }

    const __bf16* kbase = Kb + (size_t)head * S_LEN * KSTRIDE;
    const __bf16* vbase = Vt + (size_t)head * HEADD * S_LEN;

    float m8[2][8], l8[2][8];
    #pragma unroll
    for (int q = 0; q < 2; ++q)
        #pragma unroll
        for (int i = 0; i < 8; ++i) { m8[q][i] = -3.0e38f; l8[q][i] = 0.0f; }
    v8f o[2][4] = {};

    for (int kb = 0; kb < S_LEN / 32; ++kb) {
        const int key0 = kb * 32;

        // ---- scores: 2 query sub-tiles x 2 key tiles; B loaded once --------
        v8f sc[2][2] = {};
        #pragma unroll
        for (int c = 0; c < NA; ++c) {
            #pragma unroll
            for (int ntile = 0; ntile < 2; ++ntile) {
                const __bf16* kr = kbase +
                    (size_t)(key0 + ntile * 16 + col) * KSTRIDE + c * 32 + part * 16;
                v16bf b = ld_b16(kr);
                sc[0][ntile] = wmma_bf16(aq[0][c], b, sc[0][ntile]);
                sc[1][ntile] = wmma_bf16(aq[1][c], b, sc[1][ntile]);
            }
        }

        // ---- online softmax per query sub-tile -----------------------------
        #pragma unroll
        for (int q = 0; q < 2; ++q) {
            float f0[8], f1[8], p0[8], p1[8];
            #pragma unroll
            for (int i = 0; i < 8; ++i) { f0[i] = sc[q][0][i]; f1[i] = sc[q][1][i]; }
            #pragma unroll
            for (int i = 0; i < 8; ++i) {
                float cand = fmaxf(f0[i], f1[i]);
                #pragma unroll
                for (int off = 1; off < 16; off <<= 1)
                    cand = fmaxf(cand, __shfl_xor(cand, off, 32));
                float mnew  = fmaxf(m8[q][i], cand);
                float alpha = __expf(m8[q][i] - mnew);
                m8[q][i] = mnew;
                p0[i] = __expf(f0[i] - mnew);
                p1[i] = __expf(f1[i] - mnew);
                float rs = p0[i] + p1[i];
                #pragma unroll
                for (int off = 1; off < 16; off <<= 1)
                    rs += __shfl_xor(rs, off, 32);
                l8[q][i] = l8[q][i] * alpha + rs;
                #pragma unroll
                for (int j = 0; j < 4; ++j) o[q][j][i] *= alpha;
            }
            // P -> LDS (rows q*16 .. q*16+15)
            #pragma unroll
            for (int i = 0; i < 8; ++i) {
                pl[q * 16 + part * 8 + i][col]      = (__bf16)p0[i];
                pl[q * 16 + part * 8 + i][16 + col] = (__bf16)p1[i];
            }
        }
        __syncthreads();
        v16bf pa0 = ld_a2(&pl[col][part * 16],      &pl[col][part * 16 + 8]);
        v16bf pa1 = ld_a2(&pl[16 + col][part * 16], &pl[16 + col][part * 16 + 8]);
        __syncthreads();

        // ---- accumulate P @ V; V tile loaded once per j --------------------
        #pragma unroll
        for (int j = 0; j < 4; ++j) {
            const __bf16* vr =
                vbase + (size_t)(j * 16 + col) * S_LEN + key0 + part * 16;
            v16bf bv = ld_b16(vr);
            o[0][j] = wmma_bf16(pa0, bv, o[0][j]);
            o[1][j] = wmma_bf16(pa1, bv, o[1][j]);
        }
    }

    // ---- normalize + store into concat (cols head*64 .. head*64+63) --------
    const int colOfs = head * HEADD;
    #pragma unroll
    for (int q = 0; q < 2; ++q) {
        #pragma unroll
        for (int j = 0; j < 4; ++j) {
            #pragma unroll
            for (int i = 0; i < 8; ++i) {
                int row = qt * 32 + q * 16 + part * 8 + i;
                float inv = 1.0f / l8[q][i];
                concat[(size_t)row * DIMM + colOfs + j * 16 + col] =
                    (__bf16)(o[q][j][i] * inv);
            }
        }
    }
}

// ---------------------------------------------------------------------------
// Host launcher
// ---------------------------------------------------------------------------
extern "C" void kernel_launch(void* const* d_in, const int* in_sizes, int n_in,
                              void* d_out, int out_size, void* d_ws, size_t ws_size,
                              hipStream_t stream) {
    (void)in_sizes; (void)n_in; (void)out_size; (void)ws_size;

    const float* x      = (const float*)d_in[0];
    const float* qs_W   = (const float*)d_in[1];
    const float* qs_b   = (const float*)d_in[2];
    const float* ln_g   = (const float*)d_in[3];
    const float* ln_b   = (const float*)d_in[4];
    const float* q_freq = (const float*)d_in[5];
    const float* q_shift= (const float*)d_in[6];
    const float* Wq     = (const float*)d_in[7];
    const float* Wk     = (const float*)d_in[8];
    const float* Wv     = (const float*)d_in[9];
    const float* Wo     = (const float*)d_in[10];
    const float* bo     = (const float*)d_in[11];
    float* out          = (float*)d_out;

    // workspace carve-up (256B aligned)
    char*  ws  = (char*)d_ws;
    size_t off = 0;
    auto carve = [&](size_t bytes) {
        void* p = ws + off;
        off = (off + bytes + 255) & ~(size_t)255;
        return p;
    };
    const size_t WN = (size_t)THEADS * HEADD;         // 896
    __bf16* xbf    = (__bf16*)carve((size_t)S_LEN * DIMM * 2);
    __bf16* Wqkvbf = (__bf16*)carve(3 * WN * DIMM * 2);  // [Wq;Wk;Wv] merged
    __bf16* Wobf   = (__bf16*)carve((size_t)DIMM * DIMM * 2);
    __bf16* Qb     = (__bf16*)carve((size_t)NHEADS * S_LEN * KSTRIDE * 2);
    __bf16* Kb     = (__bf16*)carve((size_t)NHEADS * S_LEN * KSTRIDE * 2);
    __bf16* Vt     = (__bf16*)carve((size_t)NHEADS * HEADD * S_LEN * 2);
    __bf16* concat = (__bf16*)carve((size_t)S_LEN * DIMM * 2);

    // 1) convert operands to bf16 (QKV weights into one contiguous B operand)
    {
        int n = S_LEN * DIMM;
        cvt_f32_bf16<<<(n + 255) / 256, 256, 0, stream>>>(x, xbf, n);
        n = (int)(WN * DIMM);
        cvt_f32_bf16<<<(n + 255) / 256, 256, 0, stream>>>(Wq, Wqkvbf, n);
        cvt_f32_bf16<<<(n + 255) / 256, 256, 0, stream>>>(Wk, Wqkvbf + WN * DIMM, n);
        cvt_f32_bf16<<<(n + 255) / 256, 256, 0, stream>>>(Wv, Wqkvbf + 2 * WN * DIMM, n);
        n = DIMM * DIMM;
        cvt_f32_bf16<<<(n + 255) / 256, 256, 0, stream>>>(Wo, Wobf, n);
    }

    // 2) quantum head features (amp/phase -> cos/sin factorization)
    quantum_prep<<<dim3(S_LEN, QHEADS), 128, 0, stream>>>(
        x, qs_W, qs_b, ln_g, ln_b, q_freq, q_shift, Qb, Kb, Vt);

    // 3) fused Q/K/V projection for traditional heads (one GEMM, N = 2688)
    gemm_bf16_nt<<<dim3(S_LEN / 32, (3 * (int)WN) / 64), 32, 0, stream>>>(
        xbf, Wqkvbf, DIMM, 5, nullptr,
        Qb + (size_t)QHEADS * S_LEN * KSTRIDE,
        Kb + (size_t)QHEADS * S_LEN * KSTRIDE,
        Vt + (size_t)QHEADS * HEADD * S_LEN,
        nullptr, 0);

    // 4) flash attention: quantum heads (K-dim 128), traditional (K-dim 64)
    flash_attn<4><<<dim3(S_LEN / 32, QHEADS), 32, 0, stream>>>(
        Qb, Kb, Vt, concat, 0);
    flash_attn<2><<<dim3(S_LEN / 32, THEADS), 32, 0, stream>>>(
        Qb, Kb, Vt, concat, QHEADS);

    // 5) output projection: out = concat @ Wo^T + bo (f32)
    gemm_bf16_nt<<<dim3(S_LEN / 32, DIMM / 64), 32, 0, stream>>>(
        concat, Wobf, DIMM, 4, bo, nullptr, nullptr, nullptr, out, DIMM);
}
